// MVDNetRRPN_30416958390505
// MI455X (gfx1250) — compile-verified, matrix operand order checked by reading, not run on hardware
//
#include <hip/hip_runtime.h>
#include <hip/hip_bf16.h>

#define A_CNT   96000
#define TOPK    1000
#define NEG_INF (-__builtin_inff())

typedef _Float16 v16h __attribute__((ext_vector_type(16)));
typedef _Float16 v8h  __attribute__((ext_vector_type(8)));
typedef float    v8f  __attribute__((ext_vector_type(8)));

__device__ inline bool finf(float x) { return x == x && fabsf(x) != __builtin_inff(); }

// ---------------------------------------------------------------------------
// Rotated-box corners (matches reference _box_corners)
// ---------------------------------------------------------------------------
__device__ inline void box_corners(const float b[5], float px[4], float py[4]) {
    float t = b[4] * 0.017453292519943295f;
    float s, c;
    __sincosf(t, &s, &c);
    const float hx[4] = {0.5f, -0.5f, -0.5f, 0.5f};
    const float hy[4] = {0.5f, 0.5f, -0.5f, -0.5f};
#pragma unroll
    for (int k = 0; k < 4; k++) {
        float dx = hx[k] * b[2], dy = hy[k] * b[3];
        px[k] = b[0] + c * dx - s * dy;
        py[k] = b[1] + s * dx + c * dy;
    }
}

// ---------------------------------------------------------------------------
// Exact rotated IoU: Sutherland-Hodgman clip, faithful port of the reference
// (_clip_halfplane with MAXV=8, including its emit/drop semantics).
// ---------------------------------------------------------------------------
__device__ float pair_iou(const float b1[5], const float b2[5]) {
    float px[8], py[8];
    {
        float cx[4], cy[4];
        box_corners(b1, cx, cy);
#pragma unroll
        for (int k = 0; k < 4; k++) { px[k] = cx[k]; py[k] = cy[k]; }
#pragma unroll
        for (int k = 4; k < 8; k++) { px[k] = 0.f; py[k] = 0.f; }
    }
    int n = 4;
    float qx[4], qy[4];
    box_corners(b2, qx, qy);

#pragma unroll
    for (int e = 0; e < 4; e++) {
        float ax = qx[e], ay = qy[e];
        int e2 = (e + 1) & 3;
        float ex = qx[e2] - ax, ey = qy[e2] - ay;
        float ox[8], oy[8];
#pragma unroll
        for (int k = 0; k < 8; k++) { ox[k] = 0.f; oy[k] = 0.f; }
        int cnt = 0;
#pragma unroll
        for (int idx = 0; idx < 8; idx++) {
            bool valid = idx < n;
            int ni = (idx + 1 < n) ? idx + 1 : 0;
            float cxp = px[idx], cyp = py[idx];
            float nx = px[ni], ny = py[ni];
            float dc = ex * (cyp - ay) - ey * (cxp - ax);
            float dn = ex * (ny - ay) - ey * (nx - ax);
            bool ic = dc >= 0.f, inn = dn >= 0.f;
            float den = dc - dn;
            if (fabsf(den) < 1e-12f) den = 1e-12f;
            float t = dc / den;
            float ixp = cxp + t * (nx - cxp);
            float iyp = cyp + t * (ny - cyp);
            if (valid && ic) {
                if (cnt < 8) { ox[cnt] = cxp; oy[cnt] = cyp; }
                cnt++;
            }
            if (valid && (ic != inn)) {
                if (cnt < 8) { ox[cnt] = ixp; oy[cnt] = iyp; }
                cnt++;
            }
        }
#pragma unroll
        for (int k = 0; k < 8; k++) { px[k] = ox[k]; py[k] = oy[k]; }
        n = cnt;
    }
    float s2 = 0.f;
#pragma unroll
    for (int idx = 0; idx < 8; idx++) {
        if (idx < n) {
            int ni = (idx + 1 < n) ? idx + 1 : 0;
            s2 += px[idx] * py[ni] - py[idx] * px[ni];
        }
    }
    float inter = 0.5f * fabsf(s2);
    float a1 = b1[2] * b1[3], a2 = b2[2] * b2[3];
    return inter / fmaxf(a1 + a2 - inter, 1e-12f);
}

// ---------------------------------------------------------------------------
// Kernel 1: exact top-k with persistent per-thread stripe maxima.
// Removing the round winner invalidates exactly one thread's stripe max, so
// only that thread rescans its ~94-element stripe; everyone else reuses the
// cached candidate. Global traffic ~= one full pass + 1000 stripe rescans
// (vs 1000 full passes). Tie-break: lowest index (matches jax.lax.top_k).
// ---------------------------------------------------------------------------
__global__ void __launch_bounds__(1024) topk_kernel(
    const float* __restrict__ props_r, const float* __restrict__ log_r,
    const float* __restrict__ props_l, const float* __restrict__ log_l,
    float* __restrict__ selScore, float* __restrict__ selBox) {
    int list = blockIdx.x, img = list >> 1, feat = list & 1;
    const float* lg = (feat ? log_l : log_r) + (size_t)img * A_CNT;
    const float* pr = (feat ? props_l : props_r) + (size_t)img * A_CNT * 5;
    int tid = threadIdx.x;

    __shared__ unsigned taken[(A_CNT + 31) / 32];
    __shared__ float candV[1024];
    __shared__ int   candI[1024];
    __shared__ float rv[1024];
    __shared__ int   ri[1024];
    __shared__ int   selI[TOPK];
    __shared__ float selV[TOPK];

    for (int j = tid; j < (A_CNT + 31) / 32; j += 1024) taken[j] = 0u;

    // Initial stripe scan (stripe = indices j == tid mod 1024, ascending).
    {
        float bv = NEG_INF;
        int bi = 0x7fffffff;
        for (int j = tid; j < A_CNT; j += 1024) {
            float v = lg[j];
            if (v > bv) { bv = v; bi = j; }  // first max wins ties
        }
        candV[tid] = bv; candI[tid] = bi;
    }
    __syncthreads();

    for (int r = 0; r < TOPK; r++) {
        rv[tid] = candV[tid]; ri[tid] = candI[tid];
        __syncthreads();
        for (int off = 512; off > 0; off >>= 1) {
            if (tid < off) {
                float v2 = rv[tid + off]; int i2 = ri[tid + off];
                if (v2 > rv[tid] || (v2 == rv[tid] && i2 < ri[tid])) { rv[tid] = v2; ri[tid] = i2; }
            }
            __syncthreads();
        }
        int wi = ri[0];
        if (tid == 0) { selI[r] = wi; selV[r] = rv[0]; }
        if (tid == (wi & 1023)) {  // single writer per round; barriers separate rounds
            taken[wi >> 5] |= 1u << (wi & 31);
            float bv = NEG_INF;
            int bi = 0x7fffffff;
            for (int j = tid; j < A_CNT; j += 1024) {
                if (!((taken[j >> 5] >> (j & 31)) & 1u)) {
                    float v = lg[j];
                    if (v > bv) { bv = v; bi = j; }
                }
            }
            candV[tid] = bv; candI[tid] = bi;
        }
        __syncthreads();
    }

    // Epilogue: gather, clip, validate (reference _clip_boxes + MIN_SIZE).
    size_t o = (size_t)list * 1024 + tid;
    float b[5] = {0, 0, 0, 0, 0};
    float s = NEG_INF;
    if (tid < TOPK) {
        int src = selI[tid];
        float sv = selV[tid];
#pragma unroll
        for (int k = 0; k < 5; k++) b[k] = pr[(size_t)src * 5 + k];
        bool fin = finf(sv);
#pragma unroll
        for (int k = 0; k < 5; k++) fin = fin && finf(b[k]);
        float cx = b[0], cy = b[1], wd = b[2], ht = b[3], an = b[4];
        float x1 = fminf(fmaxf(cx - 0.5f * wd, 0.f), 320.f);
        float y1 = fminf(fmaxf(cy - 0.5f * ht, 0.f), 320.f);
        float x2 = fminf(fmaxf(cx + 0.5f * wd, 0.f), 320.f);
        float y2 = fminf(fmaxf(cy + 0.5f * ht, 0.f), 320.f);
        bool near0 = fabsf(an) <= 1.0f;
        b[0] = near0 ? 0.5f * (x1 + x2) : cx;
        b[1] = near0 ? 0.5f * (y1 + y2) : cy;
        b[2] = near0 ? (x2 - x1) : wd;
        b[3] = near0 ? (y2 - y1) : ht;
        bool valid = fin && (b[2] > 0.f) && (b[3] > 0.f);
        s = valid ? sv : NEG_INF;
    }
    selScore[o] = s;
#pragma unroll
    for (int k = 0; k < 5; k++) selBox[o * 5 + k] = b[k];
}

// ---------------------------------------------------------------------------
// Kernel 2: stable descending sort (rank-based, O(n^2) in LDS) + build f16
// bounding-circle features for the WMMA prefilter.
//  g_ij = (r_i+r_j)^2 - dist^2 = fa(i) . fb(j)   (coords scaled by 1/64)
// ---------------------------------------------------------------------------
__global__ void __launch_bounds__(1024) sort_feat_kernel(
    const float* __restrict__ inScore, const float* __restrict__ inBox,
    float* __restrict__ outScore, float* __restrict__ outBox,
    _Float16* __restrict__ fa, _Float16* __restrict__ fb) {
    int list = blockIdx.x;
    size_t base = (size_t)list * 1024;
    int tid = threadIdx.x;
    __shared__ float sc[1024];
    __shared__ float oS[1024];
    __shared__ float oB[1024 * 5];

    float myS = (tid < 1000) ? inScore[base + tid] : NEG_INF;
    float myB[5] = {0, 0, 0, 0, 0};
    if (tid < 1000) {
#pragma unroll
        for (int k = 0; k < 5; k++) myB[k] = inBox[(base + tid) * 5 + k];
    }
    sc[tid] = myS;
    __syncthreads();

    int rank = 0;
    for (int j = 0; j < 1024; j++) {
        float v = sc[j];
        rank += (v > myS) || (v == myS && j < tid);  // stable
    }
    oS[rank] = myS;
#pragma unroll
    for (int k = 0; k < 5; k++) oB[rank * 5 + k] = myB[k];
    __syncthreads();

    float s2 = oS[tid];
    float b2[5];
#pragma unroll
    for (int k = 0; k < 5; k++) b2[k] = oB[tid * 5 + k];
    outScore[base + tid] = s2;
#pragma unroll
    for (int k = 0; k < 5; k++) outBox[(base + tid) * 5 + k] = b2[k];

    const float Sg = 0.015625f;  // 1/64 keeps everything well inside f16 range
    float r = 0.5f * sqrtf(b2[2] * b2[2] + b2[3] * b2[3]);
    float xs = b2[0] * Sg, ys = b2[1] * Sg, rs = r * Sg;
    float q = rs * rs - (xs * xs + ys * ys);
    _Float16* pa = fa + (base + tid) * 8;
    _Float16* pb = fb + (base + tid) * 8;
    pa[0] = (_Float16)(2.f * xs); pa[1] = (_Float16)(2.f * ys);
    pa[2] = (_Float16)q;          pa[3] = (_Float16)1.f;
    pa[4] = (_Float16)(2.f * rs); pa[5] = (_Float16)0.f;
    pa[6] = (_Float16)0.f;        pa[7] = (_Float16)0.f;
    pb[0] = (_Float16)xs;         pb[1] = (_Float16)ys;
    pb[2] = (_Float16)1.f;        pb[3] = (_Float16)q;
    pb[4] = (_Float16)rs;         pb[5] = (_Float16)0.f;
    pb[6] = (_Float16)0.f;        pb[7] = (_Float16)0.f;
}

// ---------------------------------------------------------------------------
// Kernel 3: WMMA prefilter, fused with threshold + wave32 ballot so only a
// packed candidate BITMASK ever leaves the wave (no 4 MB f32 G round-trip).
// One wave handles 4 consecutive tj tiles, reusing the A fragment; below-
// diagonal tiles culled with wave-uniform branches. For accumulator row g,
// ballot(acc[g] >= -0.5) bits[15:0] are row ti*16+g, bits[31:16] row ti*16+8+g
// (per the CDNA5 16x16 f32 C/D layout). cand is u16 per (row, 16-col tile).
// ---------------------------------------------------------------------------
__global__ void __launch_bounds__(32) wmma_cand_kernel(
    const _Float16* __restrict__ fa, const _Float16* __restrict__ fb,
    unsigned short* __restrict__ cand) {
    int blk = blockIdx.x;  // 64 ti * 16 tj-groups = 1024
    int list = blockIdx.y;
    int ti = blk >> 4, tjg = blk & 15;
    if (tjg * 4 + 3 < ti) return;  // whole group strictly below diagonal
    int lane = threadIdx.x;
    int c = lane & 15, h = lane >> 4;

    const v8h* pa = (const v8h*)(fa + ((size_t)list * 1024 + (size_t)(ti * 16 + c)) * 8);
    v8h ta = *pa;
    v16h A;
#pragma unroll
    for (int k = 0; k < 16; k++) A[k] = (_Float16)0;
    if (h == 0) {  // lanes 16-31 hold K=8..15/24..31 -> zero in our K=8 scheme
#pragma unroll
        for (int k = 0; k < 8; k++) A[k] = ta[k];
    }

    unsigned short* cl = cand + (size_t)list * 1024 * 64;

#pragma unroll
    for (int t = 0; t < 4; t++) {
        int tj = tjg * 4 + t;
        if (tj < ti) continue;  // wave-uniform
        const v8h* pb = (const v8h*)(fb + ((size_t)list * 1024 + (size_t)(tj * 16 + c)) * 8);
        v8h tb = *pb;
        v16h B;
#pragma unroll
        for (int k = 0; k < 16; k++) B[k] = (_Float16)0;
        if (h == 0) {
#pragma unroll
            for (int k = 0; k < 8; k++) B[k] = tb[k];
        }
        v8f acc = {0.f, 0.f, 0.f, 0.f, 0.f, 0.f, 0.f, 0.f};
        acc = __builtin_amdgcn_wmma_f32_16x16x32_f16(
            /*neg_a=*/false, A, /*neg_b=*/false, B,
            /*c_mod=*/(short)0, acc, /*reuse_a=*/false, /*reuse_b=*/false);

        unsigned bal[8];
#pragma unroll
        for (int g = 0; g < 8; g++)
            bal[g] = __builtin_amdgcn_ballot_w32(acc[g] >= -0.5f);

        if (lane < 16) {  // lanes 0-7: low half (rows g); lanes 8-15: high half (rows 8+g)
            int g = lane & 7;
            unsigned v = 0;
#pragma unroll
            for (int k = 0; k < 8; k++)
                if (g == k) v = bal[k];
            unsigned half = (lane < 8) ? (v & 0xffffu) : (v >> 16);
            int row = ti * 16 + ((lane < 8) ? g : 8 + g);
            cl[(size_t)row * 64 + tj] = (unsigned short)half;
        }
    }
}

// ---------------------------------------------------------------------------
// Kernel 4: suppression bitmask. Thread = (row i, 32-column word). Exact
// polygon IoU only where the WMMA circle test set a candidate bit. Bits with
// j <= i may be garbage (tile culled) but are never consulted.
// ---------------------------------------------------------------------------
__global__ void __launch_bounds__(256) mask_kernel(
    const float* __restrict__ srtBox, const unsigned short* __restrict__ cand,
    unsigned* __restrict__ mask) {
    int list = blockIdx.y;
    int T = blockIdx.x * 256 + threadIdx.x;  // 0..32767
    int i = T >> 5, w = T & 31;
    unsigned word = 0u;
    if (i < 1000) {
        float b1[5];
#pragma unroll
        for (int k = 0; k < 5; k++) b1[k] = srtBox[((size_t)list * 1024 + i) * 5 + k];
        const unsigned short* cl = cand + (size_t)list * 1024 * 64 + (size_t)i * 64;
        unsigned cw = (unsigned)cl[2 * w] | ((unsigned)cl[2 * w + 1] << 16);
        for (int b = 0; b < 32; b++) {
            int j = (w << 5) + b;
            if (j > i && j < 1000 && ((cw >> b) & 1u)) {
                float b2[5];
#pragma unroll
                for (int k = 0; k < 5; k++) b2[k] = srtBox[((size_t)list * 1024 + j) * 5 + k];
                if (pair_iou(b1, b2) > 0.7f) word |= (1u << b);
            }
        }
    }
    mask[((size_t)list * 1024 + (size_t)i) * 32 + w] = word;
}

// ---------------------------------------------------------------------------
// Kernel 5a: sequential greedy NMS + compaction into concat buffers (keep 500).
// ---------------------------------------------------------------------------
__global__ void __launch_bounds__(128) compactA_kernel(
    const float* __restrict__ srtScore, const float* __restrict__ srtBox,
    const unsigned* __restrict__ mask,
    float* __restrict__ catScore, float* __restrict__ catBox) {
    int list = blockIdx.x, img = list >> 1, feat = list & 1;
    int tid = threadIdx.x;
    __shared__ float sc[1024];
    __shared__ unsigned sup[32];
    __shared__ int selIdx[1024];
    __shared__ int cnt;
    size_t base = (size_t)list * 1024;

    for (int j = tid; j < 1024; j += 128) sc[j] = srtScore[base + j];
    if (tid == 0) cnt = 0;
    __syncthreads();
    if (tid < 32) {
        unsigned wd = 0u;
#pragma unroll
        for (int b = 0; b < 32; b++) {
            int j = tid * 32 + b;
            if (!(sc[j] > NEG_INF)) wd |= 1u << b;  // suppressed init = ~valid
        }
        sup[tid] = wd;
    }
    __syncthreads();

    for (int i = 0; i < 1000; i++) {
        bool kept = (sc[i] > NEG_INF) && !((sup[i >> 5] >> (i & 31)) & 1u);
        if (kept) {
            if (tid < 32) sup[tid] |= mask[(base + i) * 32 + tid];  // only bits j>i
            if (tid == 0) { selIdx[cnt] = i; cnt = cnt + 1; }
        }
        __syncthreads();
    }

    int total = cnt;
    size_t dbase = (size_t)img * 1024 + (size_t)feat * 500;
    for (int t = tid; t < 500; t += 128) {
        float bx[5] = {0, 0, 0, 0, 0};
        float s = NEG_INF;
        if (t < total) {
            int si = selIdx[t];
#pragma unroll
            for (int k = 0; k < 5; k++) bx[k] = srtBox[(base + si) * 5 + k];
            s = sc[si];
        }
        catScore[dbase + t] = s;
#pragma unroll
        for (int k = 0; k < 5; k++) catBox[(dbase + t) * 5 + k] = bx[k];
    }
}

// ---------------------------------------------------------------------------
// Kernel 5b: final NMS + emit (2,1000,6) rows [cx,cy,w,h,ang,score].
// ---------------------------------------------------------------------------
__global__ void __launch_bounds__(128) compactF_kernel(
    const float* __restrict__ srtScore, const float* __restrict__ srtBox,
    const unsigned* __restrict__ mask, float* __restrict__ out) {
    int img = blockIdx.x;
    int tid = threadIdx.x;
    __shared__ float sc[1024];
    __shared__ unsigned sup[32];
    __shared__ int selIdx[1024];
    __shared__ int cnt;
    size_t base = (size_t)img * 1024;

    for (int j = tid; j < 1024; j += 128) sc[j] = srtScore[base + j];
    if (tid == 0) cnt = 0;
    __syncthreads();
    if (tid < 32) {
        unsigned wd = 0u;
#pragma unroll
        for (int b = 0; b < 32; b++) {
            int j = tid * 32 + b;
            if (!(sc[j] > NEG_INF)) wd |= 1u << b;
        }
        sup[tid] = wd;
    }
    __syncthreads();

    for (int i = 0; i < 1000; i++) {
        bool kept = (sc[i] > NEG_INF) && !((sup[i >> 5] >> (i & 31)) & 1u);
        if (kept) {
            if (tid < 32) sup[tid] |= mask[(base + i) * 32 + tid];
            if (tid == 0) { selIdx[cnt] = i; cnt = cnt + 1; }
        }
        __syncthreads();
    }

    int total = cnt;
    for (int t = tid; t < 1000; t += 128) {
        float bx[5] = {0, 0, 0, 0, 0};
        float s = NEG_INF;
        if (t < total) {
            int si = selIdx[t];
#pragma unroll
            for (int k = 0; k < 5; k++) bx[k] = srtBox[(base + si) * 5 + k];
            s = sc[si];
        }
        float* row = out + (size_t)img * 6000 + (size_t)t * 6;
#pragma unroll
        for (int k = 0; k < 5; k++) row[k] = bx[k];
        row[5] = s;
    }
}

// ---------------------------------------------------------------------------
// Orchestration. Workspace ~1 MB: the f32 G matrix is gone — the WMMA kernel
// emits packed candidate bits directly (128 KB/list).
// ---------------------------------------------------------------------------
extern "C" void kernel_launch(void* const* d_in, const int* in_sizes, int n_in,
                              void* d_out, int out_size, void* d_ws, size_t ws_size,
                              hipStream_t stream) {
    (void)in_sizes; (void)n_in; (void)out_size; (void)ws_size;
    const float* props_r = (const float*)d_in[0];
    const float* log_r   = (const float*)d_in[1];
    const float* props_l = (const float*)d_in[2];
    const float* log_l   = (const float*)d_in[3];

    char* w = (char*)d_ws;
    auto carve = [&](size_t bytes) -> void* {
        void* p = (void*)w;
        w += (bytes + 255) & ~(size_t)255;
        return p;
    };
    float*          selScore  = (float*)carve((size_t)4 * 1024 * 4);
    float*          selBox    = (float*)carve((size_t)4 * 1024 * 5 * 4);
    float*          srtScoreA = (float*)carve((size_t)4 * 1024 * 4);
    float*          srtBoxA   = (float*)carve((size_t)4 * 1024 * 5 * 4);
    _Float16*       faA       = (_Float16*)carve((size_t)4 * 1024 * 8 * 2);
    _Float16*       fbA       = (_Float16*)carve((size_t)4 * 1024 * 8 * 2);
    float*          catScore  = (float*)carve((size_t)2 * 1024 * 4);
    float*          catBox    = (float*)carve((size_t)2 * 1024 * 5 * 4);
    float*          srtScoreB = (float*)carve((size_t)2 * 1024 * 4);
    float*          srtBoxB   = (float*)carve((size_t)2 * 1024 * 5 * 4);
    _Float16*       faB       = (_Float16*)carve((size_t)2 * 1024 * 8 * 2);
    _Float16*       fbB       = (_Float16*)carve((size_t)2 * 1024 * 8 * 2);
    unsigned*       maskBuf   = (unsigned*)carve((size_t)4 * 1024 * 32 * 4);
    unsigned short* cand      = (unsigned short*)carve((size_t)4 * 1024 * 64 * 2);

    // Stage A: per (image, feature) — 4 lists in parallel per launch.
    topk_kernel<<<4, 1024, 0, stream>>>(props_r, log_r, props_l, log_l, selScore, selBox);
    sort_feat_kernel<<<4, 1024, 0, stream>>>(selScore, selBox, srtScoreA, srtBoxA, faA, fbA);
    wmma_cand_kernel<<<dim3(1024, 4), 32, 0, stream>>>(faA, fbA, cand);
    mask_kernel<<<dim3(128, 4), 256, 0, stream>>>(srtBoxA, cand, maskBuf);
    compactA_kernel<<<4, 128, 0, stream>>>(srtScoreA, srtBoxA, maskBuf, catScore, catBox);

    // Stage B: final per-image NMS over radar(500)+lidar(500). Reuse cand/mask.
    sort_feat_kernel<<<2, 1024, 0, stream>>>(catScore, catBox, srtScoreB, srtBoxB, faB, fbB);
    wmma_cand_kernel<<<dim3(1024, 2), 32, 0, stream>>>(faB, fbB, cand);
    mask_kernel<<<dim3(128, 2), 256, 0, stream>>>(srtBoxB, cand, maskBuf);
    compactF_kernel<<<2, 128, 0, stream>>>(srtScoreB, srtBoxB, maskBuf, (float*)d_out);
}
